// MoEGate_85392539779349
// MI455X (gfx1250) — compile-verified, matrix operand order checked by reading, not run on hardware
//
#include <hip/hip_runtime.h>

typedef __attribute__((ext_vector_type(16))) __bf16 v16bf;
typedef __attribute__((ext_vector_type(8)))  float  v8f;

union Frag {
  unsigned int u[8];   // 8 dwords = 16 packed bf16
  v16bf b;
};

// Truncation split: hi = f's top 16 bits (valid bf16), lo = exact residual
// truncated to bf16. hi+lo reproduces f to ~2^-17 relative; cheap VALU only.
// v_perm_b32 packs the two high-16s of a float pair in one instruction.
__device__ __forceinline__ void split4(float4 v,
    unsigned int& h01, unsigned int& h23,
    unsigned int& l01, unsigned int& l23) {
  unsigned int u0 = __float_as_uint(v.x), u1 = __float_as_uint(v.y);
  unsigned int u2 = __float_as_uint(v.z), u3 = __float_as_uint(v.w);
  h01 = __builtin_amdgcn_perm(u1, u0, 0x07060302u);  // {u1.hi16, u0.hi16}
  h23 = __builtin_amdgcn_perm(u3, u2, 0x07060302u);
  float r0 = v.x - __uint_as_float(u0 & 0xffff0000u); // exact residuals
  float r1 = v.y - __uint_as_float(u1 & 0xffff0000u);
  float r2 = v.z - __uint_as_float(u2 & 0xffff0000u);
  float r3 = v.w - __uint_as_float(u3 & 0xffff0000u);
  l01 = __builtin_amdgcn_perm(__float_as_uint(r1), __float_as_uint(r0), 0x07060302u);
  l23 = __builtin_amdgcn_perm(__float_as_uint(r3), __float_as_uint(r2), 0x07060302u);
}

constexpr int D  = 2048;   // hidden dim
constexpr int E  = 64;     // experts
constexpr int BT = 128;    // tokens per block (8 waves x 16)
constexpr int KC = 128;    // K chunk staged in LDS
constexpr int TOPK = 8;
constexpr int LGS = E + 1; // logits LDS row stride (bank-conflict-free)

__global__ __launch_bounds__(256)
void moe_gate_kernel(const float* __restrict__ x, const float* __restrict__ w,
                     float* __restrict__ out_vals, int* __restrict__ out_idx) {
  // Weight stage (32 KB) and logits (33 KB) are live in disjoint phases:
  // overlap them so total LDS stays ~33 KB (occupancy + CU-mode safe).
  __shared__ union {
    struct {
      alignas(16) unsigned int hi[E * KC / 2];  // packed bf16 hi
      alignas(16) unsigned int lo[E * KC / 2];  // packed bf16 lo
    } wgt;
    alignas(16) float lg[BT * LGS];
  } sm;

  const int tid  = threadIdx.x;
  const int wv   = tid >> 5;      // wave 0..7
  const int lane = tid & 31;
  const int half = lane >> 4;     // lane group
  const int l16  = lane & 15;
  const int tblk = blockIdx.x * BT;
  const float* xrow = x + (size_t)(tblk + wv * 16 + l16) * D;

  v8f acc[4] = {{}, {}, {}, {}};  // 4 tiles of 16 experts, 16x16 f32 each

  for (int kc = 0; kc < D; kc += KC) {
    // ---- cooperative stage: weight chunk [64][KC] fp32 -> bf16 hi/lo in LDS
    #pragma unroll
    for (int i = 0; i < (E * KC / 4) / 256; ++i) {   // 8 float4 per thread
      int f  = tid + i * 256;                        // float4 id 0..2047
      int e  = f >> 5;                               // 32 float4 per expert row
      int jc = (f & 31) << 2;                        // k offset in chunk
      float4 wf = *(const float4*)(w + (size_t)e * D + kc + jc);
      unsigned int h01, h23, l01, l23;
      split4(wf, h01, h23, l01, l23);
      int o = e * (KC / 2) + (jc >> 1);
      sm.wgt.hi[o] = h01; sm.wgt.hi[o + 1] = h23;
      sm.wgt.lo[o] = l01; sm.wgt.lo[o + 1] = l23;
    }
    __syncthreads();

    // ---- GEMM over this K chunk
    #pragma unroll
    for (int ks = 0; ks < KC; ks += 32) {
      // A fragment (16x32 bf16 layout): k = ks + 8*half + {0..7, 16..23}
      const float* xp = xrow + kc + ks + half * 8;
      float4 a0 = *(const float4*)(xp);
      float4 a1 = *(const float4*)(xp + 4);
      float4 a2 = *(const float4*)(xp + 16);
      float4 a3 = *(const float4*)(xp + 20);
      Frag Ah, Al;
      split4(a0, Ah.u[0], Ah.u[1], Al.u[0], Al.u[1]);
      split4(a1, Ah.u[2], Ah.u[3], Al.u[2], Al.u[3]);
      split4(a2, Ah.u[4], Ah.u[5], Al.u[4], Al.u[5]);
      split4(a3, Ah.u[6], Ah.u[7], Al.u[6], Al.u[7]);

      const int bo = (ks >> 1) + half * 4;  // dword offset within expert row
      #pragma unroll
      for (int et = 0; et < 4; ++et) {
        const unsigned int* ph = sm.wgt.hi + (et * 16 + l16) * (KC / 2) + bo;
        const unsigned int* pl = sm.wgt.lo + (et * 16 + l16) * (KC / 2) + bo;
        Frag Bh, Bl;
        *(uint4*)&Bh.u[0] = *(const uint4*)(ph);
        *(uint4*)&Bh.u[4] = *(const uint4*)(ph + 8);
        *(uint4*)&Bl.u[0] = *(const uint4*)(pl);
        *(uint4*)&Bl.u[4] = *(const uint4*)(pl + 8);
        // fp32-accurate product: hi*hi + lo*hi + hi*lo (fp32 accumulate)
        acc[et] = __builtin_amdgcn_wmma_f32_16x16x32_bf16(
            false, Ah.b, false, Bh.b, (short)0, acc[et], false, false);
        acc[et] = __builtin_amdgcn_wmma_f32_16x16x32_bf16(
            false, Al.b, false, Bh.b, (short)0, acc[et], false, false);
        acc[et] = __builtin_amdgcn_wmma_f32_16x16x32_bf16(
            false, Ah.b, false, Bl.b, (short)0, acc[et], false, false);
      }
    }
    __syncthreads();
  }

  // ---- dump logits: C layout: vgpr j -> token half*8+j, lane l16 -> expert
  #pragma unroll
  for (int et = 0; et < 4; ++et) {
    #pragma unroll
    for (int j = 0; j < 8; ++j) {
      sm.lg[(wv * 16 + half * 8 + j) * LGS + et * 16 + l16] = acc[et][j];
    }
  }
  __syncthreads();

  // ---- softmax + top-8 (one thread per token)
  if (tid < BT) {
    float* lg = sm.lg + tid * LGS;
    float m = -3.402823466e38f;
    for (int e = 0; e < E; ++e) m = fmaxf(m, lg[e]);
    float denom = 0.f;
    for (int e = 0; e < E; ++e) denom += __expf(lg[e] - m);
    const int tg = tblk + tid;
    for (int p = 0; p < TOPK; ++p) {
      float best = -3.402823466e38f; int bi = 0;
      for (int e = 0; e < E; ++e) {
        float s = lg[e];
        if (s > best) { best = s; bi = e; }  // strict: lowest index wins ties
      }
      out_vals[tg * TOPK + p] = __expf(best - m) / denom;
      out_idx[tg * TOPK + p]  = bi;
      lg[bi] = -3.402823466e38f;
    }
  }
}

extern "C" void kernel_launch(void* const* d_in, const int* in_sizes, int n_in,
                              void* d_out, int out_size, void* d_ws, size_t ws_size,
                              hipStream_t stream) {
  const float* x = (const float*)d_in[0];
  const float* w = (const float*)d_in[1];
  const int ntok = in_sizes[0] / D;          // 32768
  float* out_vals = (float*)d_out;           // first ntok*8 floats
  int*   out_idx  = (int*)d_out + (size_t)ntok * TOPK;  // then ntok*8 int32
  dim3 grid(ntok / BT), block(256);
  hipLaunchKernelGGL(moe_gate_kernel, grid, block, 0, stream,
                     x, w, out_vals, out_idx);
}
